// Network_47854525612304
// MI455X (gfx1250) — matrix operand order repaired and checked by reading, not compiled, not run on hardware
//
#include <hip/hip_runtime.h>
#include <hip/hip_bf16.h>
#include <math.h>

typedef __attribute__((ext_vector_type(16))) _Float16 v16h;
typedef __attribute__((ext_vector_type(8)))  float    v8f;

#define N_ROWS 65536
#define DIM    1000
#define EPSF   1e-8f

// ---------------------------------------------------------------------------
// Pass 1: cosine similarity. One wave32 per row. HBM-bound: 262MB @ 23.3TB/s.
// Reads E row + q as float4 (b128), accumulates dot, |e|^2, |q|^2 in one pass.
// Writes sims into d_out (kernel 2 consumes then overwrites the same slots).
// ---------------------------------------------------------------------------
__global__ __launch_bounds__(256) void sims_kernel(
    const float* __restrict__ E, const float* __restrict__ q,
    float* __restrict__ sims)
{
    const int wave = (blockIdx.x * blockDim.x + threadIdx.x) >> 5;
    const int lane = threadIdx.x & 31;

    const float4* __restrict__ Er = (const float4*)(E + (size_t)wave * DIM);
    const float4* __restrict__ Qv = (const float4*)q;

    float dot = 0.f, nrm = 0.f, qq = 0.f;
    // DIM=1000 floats = 250 float4 (row stride 4000B is 16B aligned)
    for (int i = lane; i < 250; i += 32) {
        float4 e = Er[i];
        float4 v = Qv[i];
        dot = fmaf(e.x, v.x, dot); dot = fmaf(e.y, v.y, dot);
        dot = fmaf(e.z, v.z, dot); dot = fmaf(e.w, v.w, dot);
        nrm = fmaf(e.x, e.x, nrm); nrm = fmaf(e.y, e.y, nrm);
        nrm = fmaf(e.z, e.z, nrm); nrm = fmaf(e.w, e.w, nrm);
        qq  = fmaf(v.x, v.x, qq);  qq  = fmaf(v.y, v.y, qq);
        qq  = fmaf(v.z, v.z, qq);  qq  = fmaf(v.w, v.w, qq);
    }
    #pragma unroll
    for (int m = 16; m >= 1; m >>= 1) {
        dot += __shfl_xor(dot, m, 32);
        nrm += __shfl_xor(nrm, m, 32);
        qq  += __shfl_xor(qq,  m, 32);
    }
    if (lane == 0) {
        float denom = sqrtf(nrm) * sqrtf(qq);
        sims[wave] = dot / fmaxf(denom, EPSF);
    }
}

// ---------------------------------------------------------------------------
// Pass 2: MLP via WMMA. 512 blocks x 8 waves; each wave owns 16 rows.
// w2 (512x64) and w3 (64x32) staged in LDS as f16 in B-fragment layout:
//   fragment(kc,nt): lane holds column n=lane&15, K = (lane>=16)*16 + h,
//   16 contiguous halfs per lane (one v16h).
// A-fragment (16-bit, 16x32): lane holds row m=lane&15,
//   K(h) = 16*(lane>=16) + h (same contiguous order as B).
// C/D layout: lane n=lane&15, VGPR r holds M = r + 8*(lane>=16).
// ---------------------------------------------------------------------------
__global__ __launch_bounds__(256) void mlp_kernel(
    const float* __restrict__ sims_in,   // == d_out, read before overwrite
    const float* __restrict__ w1, const float* __restrict__ b1,
    const float* __restrict__ w2, const float* __restrict__ b2,
    const float* __restrict__ w3, const float* __restrict__ b3,
    const float* __restrict__ w4, const float* __restrict__ b4,
    float* __restrict__ out)
{
    __shared__ v16h w2frag[16 * 4 * 32];   // [kc][nt][lane] -> 64 KB
    __shared__ v16h w3frag[2 * 2 * 32];    // [kc3][nt3][lane] -> 4 KB
    __shared__ float w1s[512];
    __shared__ float b1s[512];
    __shared__ _Float16 h2s[8][16 * 64];   // per-wave 16x64 activation scratch

    const int tid  = threadIdx.x;
    const int lane = tid & 31;
    const int wv   = tid >> 5;
    const int lidx = lane & 15;            // N-column / M-row local index
    const int g    = lane >> 4;            // lane-group (K/M half select)

    // ---- cooperative LDS staging (weights -> f16 B-fragment layout) ----
    {
        _Float16* w2h = (_Float16*)w2frag;
        for (int i = tid; i < 16 * 4 * 32 * 16; i += 256) {
            int h  = i & 15;
            int f  = i >> 4;
            int ln = f & 31;
            int nk = f >> 5;               // kc*4 + nt
            int nt = nk & 3;
            int kc = nk >> 2;
            int k  = kc * 32 + (ln >> 4) * 16 + h;
            int n  = nt * 16 + (ln & 15);
            w2h[i] = (_Float16)w2[k * 64 + n];
        }
        _Float16* w3h = (_Float16*)w3frag;
        for (int i = tid; i < 2 * 2 * 32 * 16; i += 256) {
            int h  = i & 15;
            int f  = i >> 4;
            int ln = f & 31;
            int nk = f >> 5;               // kc3*2 + nt3
            int nt = nk & 1;
            int kc = nk >> 1;
            int k  = kc * 32 + (ln >> 4) * 16 + h;
            int n  = nt * 16 + (ln & 15);
            w3h[i] = (_Float16)w3[k * 32 + n];
        }
        for (int i = tid; i < 512; i += 256) { w1s[i] = w1[i]; b1s[i] = b1[i]; }
    }
    __syncthreads();

    const int tile = blockIdx.x * 8 + wv;  // 4096 tiles of 16 rows
    const int base = tile * 16;

    // scalar input for this lane's A-row (m = lidx)
    const float s = sims_in[base + lidx];

    // ---- layer 2: (16x512) x (512x64), f16 WMMA, f32 accumulate ----
    v8f acc0 = {}, acc1 = {}, acc2 = {}, acc3v = {};
    #pragma unroll
    for (int kc = 0; kc < 16; ++kc) {
        // build A fragment: h1 = relu(s*w1 + b1), rank-1 layer fused in
        v16h a;
        #pragma unroll
        for (int h = 0; h < 16; ++h) {
            int k = kc * 32 + g * 16 + h;
            float v = fmaf(s, w1s[k], b1s[k]);
            a[h] = (_Float16)fmaxf(v, 0.0f);
        }
        const v16h b0 = w2frag[(kc * 4 + 0) * 32 + lane];
        const v16h b1f = w2frag[(kc * 4 + 1) * 32 + lane];
        const v16h b2f = w2frag[(kc * 4 + 2) * 32 + lane];
        const v16h b3f = w2frag[(kc * 4 + 3) * 32 + lane];
        acc0  = __builtin_amdgcn_wmma_f32_16x16x32_f16(false, a, false, b0,  (short)0, acc0,  false, false);
        acc1  = __builtin_amdgcn_wmma_f32_16x16x32_f16(false, a, false, b1f, (short)0, acc1,  false, false);
        acc2  = __builtin_amdgcn_wmma_f32_16x16x32_f16(false, a, false, b2f, (short)0, acc2,  false, false);
        acc3v = __builtin_amdgcn_wmma_f32_16x16x32_f16(false, a, false, b3f, (short)0, acc3v, false, false);
    }

    // ---- bias + relu, C-layout -> LDS (row-major 16x64 f16) ----
    {
        v8f* accs[4] = { &acc0, &acc1, &acc2, &acc3v };
        #pragma unroll
        for (int nt = 0; nt < 4; ++nt) {
            const int n = nt * 16 + lidx;
            const float bias = b2[n];
            #pragma unroll
            for (int r = 0; r < 8; ++r) {
                const int m = r + 8 * g;
                float v = fmaxf((*accs[nt])[r] + bias, 0.0f);
                h2s[wv][m * 64 + n] = (_Float16)v;
            }
        }
    }
    // same-wave LDS ops are in-order: store->load RAW through h2s is safe

    // ---- layer 3: (16x64) x (64x32) ----
    v8f c0 = {}, c1 = {};
    #pragma unroll
    for (int kc = 0; kc < 2; ++kc) {
        v16h a;
        #pragma unroll
        for (int h = 0; h < 16; ++h) {
            int k = kc * 32 + g * 16 + h;
            a[h] = h2s[wv][lidx * 64 + k];   // m = lidx
        }
        const v16h bb0 = w3frag[(kc * 2 + 0) * 32 + lane];
        const v16h bb1 = w3frag[(kc * 2 + 1) * 32 + lane];
        c0 = __builtin_amdgcn_wmma_f32_16x16x32_f16(false, a, false, bb0, (short)0, c0, false, false);
        c1 = __builtin_amdgcn_wmma_f32_16x16x32_f16(false, a, false, bb1, (short)0, c1, false, false);
    }

    // ---- layer 4: relu(h3) . w4 + b4 -> sigmoid ----
    const float b3a = b3[lidx],      b3b = b3[16 + lidx];
    const float w4a = w4[lidx],      w4b = w4[16 + lidx];
    const float bb4 = b4[0];
    #pragma unroll
    for (int r = 0; r < 8; ++r) {
        float p = fmaxf(c0[r] + b3a, 0.0f) * w4a
                + fmaxf(c1[r] + b3b, 0.0f) * w4b;
        // reduce over the 16 lanes holding this row's 32 K-columns (2 per lane)
        p += __shfl_xor(p, 1, 16);
        p += __shfl_xor(p, 2, 16);
        p += __shfl_xor(p, 4, 16);
        p += __shfl_xor(p, 8, 16);
        if (lidx == 0) {
            const int m = r + 8 * g;
            float x = p + bb4;
            out[base + m] = 1.0f / (1.0f + expf(-x));
        }
    }
}

// ---------------------------------------------------------------------------
extern "C" void kernel_launch(void* const* d_in, const int* in_sizes, int n_in,
                              void* d_out, int out_size, void* d_ws, size_t ws_size,
                              hipStream_t stream)
{
    const float* query = (const float*)d_in[0];
    const float* emb   = (const float*)d_in[1];
    const float* w1    = (const float*)d_in[2];
    const float* b1    = (const float*)d_in[3];
    const float* w2    = (const float*)d_in[4];
    const float* b2    = (const float*)d_in[5];
    const float* w3    = (const float*)d_in[6];
    const float* b3    = (const float*)d_in[7];
    const float* w4    = (const float*)d_in[8];
    const float* b4    = (const float*)d_in[9];
    float* out = (float*)d_out;

    // Pass 1: 65536 waves (8 per block) -> sims written into d_out
    sims_kernel<<<N_ROWS / 8, 256, 0, stream>>>(emb, query, out);

    // Pass 2: 4096 row-tiles of 16, 8 waves/block -> 512 blocks
    mlp_kernel<<<512, 256, 0, stream>>>(out, w1, b1, w2, b2, w3, b3, w4, b4, out);
}
